// BatchTreeEncoder_40389872451852
// MI455X (gfx1250) — matrix operand order repaired and checked by reading, not compiled, not run on hardware
//
#include <hip/hip_runtime.h>
#include <math.h>

// ---------------------------------------------------------------------------
// BatchTreeEncoder for MI455X (gfx1250): bf16 WMMA GRU + attention + TDM
//   A=4 children, D=7 levels, B=64 batch, E=H=128, N_NODES=5461
// ---------------------------------------------------------------------------

typedef __attribute__((ext_vector_type(16))) __bf16 v16bf;
typedef __attribute__((ext_vector_type(8)))  __bf16 v8bf;
typedef __attribute__((ext_vector_type(8)))  float  v8f;
typedef __attribute__((ext_vector_type(4)))  unsigned int u32x4;
typedef __attribute__((ext_vector_type(8)))  int    i32x8;
typedef __attribute__((ext_vector_type(4)))  int    i32x4;

#if defined(__has_builtin)
#if __has_builtin(__builtin_amdgcn_tensor_load_to_lds)
#define HAVE_TDM 1
#endif
#endif
#if defined(__has_include)
#if __has_include(<hip/amd_detail/amd_gfx1250_TDM.h>)
#define TDM_6ARG 1   // amdgpu-toolchain / therock headers -> 6-arg builtin
#endif
#endif

#define LDA 136                    // padded LDS row (elems) -> conflict-free frag reads
#define SM_CH   0                  // ch:  256 x LDA bf16  (69632 B)
#define SM_X    69632              // x:    64 x LDA bf16  (17408 B)
#define SM_H0   87040              // h0:   64 x LDA bf16  (17408 B)
#define SM_S    104448             // s:   256 f32
#define SM_AL   105472             // alpha: 256 f32
#define SMEM_BYTES 106496

// branch-free transcendentals on native v_exp_f32 / v_rcp_f32
__device__ __forceinline__ float fast_sigmoid(float x) {
  return __builtin_amdgcn_rcpf(1.0f + __expf(-x));
}
__device__ __forceinline__ float fast_tanh(float x) {
  // 1 - 2/(e^{2x}+1): exact limits at +/-inf, no branches
  return 1.0f - 2.0f * __builtin_amdgcn_rcpf(__expf(2.0f * x) + 1.0f);
}

// float atomic max via sign-split int/uint trick (init must be -inf)
__device__ __forceinline__ void atomicMaxFloat(float* addr, float val) {
  if (val >= 0.0f) atomicMax((int*)addr, __float_as_int(val));
  else             atomicMin((unsigned int*)addr, __float_as_uint(val));
}

// A-fragment (16x32 bf16, MxK): per lane-half, K = {base..base+15}
__device__ __forceinline__ v16bf load_afrag(const __bf16* p) {
  v8bf lo = *(const v8bf*)(p);
  v8bf hi = *(const v8bf*)(p + 8);
  v16bf f;
#pragma unroll
  for (int i = 0; i < 8; ++i) { f[i] = lo[i]; f[i + 8] = hi[i]; }
  return f;
}

// 16x16 tile, K=128 accumulate.  arow already includes (m0+l)*LDA + half*16,
// brow already includes (n0+l)*128 + half*16  (B[k][n] = w[n][k], contiguous k).
__device__ __forceinline__ v8f gemm16(const __bf16* __restrict__ arow,
                                      const __bf16* __restrict__ brow, v8f acc) {
#pragma unroll
  for (int k = 0; k < 128; k += 32) {
    v16bf a = load_afrag(arow + k);
    v16bf b = *(const v16bf*)(brow + k);
    acc = __builtin_amdgcn_wmma_f32_16x16x32_bf16(false, a, false, b,
                                                  (short)0, acc, false, false);
  }
  return acc;
}

// --- prep: weights -> bf16 (sent_w transposed), init out to -inf -------------
__global__ void prep_kernel(const float* __restrict__ w_ih,
                            const float* __restrict__ w_hh,
                            const float* __restrict__ sent_w,
                            unsigned short* __restrict__ w_ih_b,
                            unsigned short* __restrict__ w_hh_b,
                            unsigned short* __restrict__ sent_wT_b,
                            float* __restrict__ out) {
  int i = blockIdx.x * 256 + threadIdx.x;
  if (i < 49152) {
    ((__bf16*)w_ih_b)[i] = (__bf16)w_ih[i];
    ((__bf16*)w_hh_b)[i] = (__bf16)w_hh[i];
  }
  if (i < 16384) {               // sent_wT[c][h] = sent_w[h][c]
    int c = i >> 7, h = i & 127;
    ((__bf16*)sent_wT_b)[i] = (__bf16)sent_w[h * 128 + c];
  }
  if (i < 8192) out[i] = -INFINITY;
}

// --- one tree level: grid = #nodes, block = 256 (8 waves) --------------------
__global__ __launch_bounds__(256)
void tree_gru_level(const int* __restrict__ tokens,
                    const float* __restrict__ emb,
                    const float* __restrict__ sent_b,
                    const float* __restrict__ ctx_w,
                    const float* __restrict__ b_ih,
                    const float* __restrict__ b_hh,
                    const unsigned short* __restrict__ w_ih_u,
                    const unsigned short* __restrict__ w_hh_u,
                    const unsigned short* __restrict__ sent_wT_u,
                    unsigned short* __restrict__ hs_u,
                    float* __restrict__ out,
                    int off, int off_child, int has_children) {
  extern __shared__ char smem[];
  __bf16* chL = (__bf16*)(smem + SM_CH);
  __bf16* xL  = (__bf16*)(smem + SM_X);
  __bf16* h0L = (__bf16*)(smem + SM_H0);
  float*  sL  = (float*)(smem + SM_S);
  float*  aL  = (float*)(smem + SM_AL);

  const __bf16* wih = (const __bf16*)w_ih_u;
  const __bf16* whh = (const __bf16*)w_hh_u;
  const __bf16* swT = (const __bf16*)sent_wT_u;
  __bf16* hs = (__bf16*)hs_u;

  const int tid  = threadIdx.x;
  const int lane = tid & 31;
  const int wave = tid >> 5;
  const int hf   = lane >> 4;       // lane half (0/1)
  const int l    = lane & 15;
  const int node = blockIdx.x;

  sL[tid] = 0.0f;

#if defined(HAVE_TDM)
  // ---- Tensor Data Mover: DMA 4 children (64KB) into LDS, row-padded to 272B,
  //      overlapped with the embedding gather below. Issued once (wave 0).
  if (has_children && wave == 0) {
    unsigned long long ga =
        (unsigned long long)(size_t)(hs + (size_t)(off_child + node * 4) * 8192);
    unsigned lds_off = (unsigned)(size_t)chL;   // flat low 32 bits = LDS offset
    u32x4 g0;
    g0[0] = 1u;                                            // count=1 (valid D#)
    g0[1] = lds_off;                                       // lds_addr
    g0[2] = (unsigned)(ga & 0xffffffffu);                  // global_addr[31:0]
    g0[3] = (unsigned)((ga >> 32) & 0x01ffffffu) | (2u << 30); // addr[56:32]|type=2
    i32x8 g1;
    g1[0] = (int)((3u << 16) |      // data_size = 8 bytes
                  (1u << 20) |      // pad_enable
                  (5u << 22) |      // pad_interval: every 64 dwords (256B)
                  (3u << 25));      // pad_amount: 4 dwords (16B) -> 272B rows
    g1[1] = (int)(8192u << 16);     // tensor_dim0[15:0]=8192 in bits[63:48]
    g1[2] = (int)(1u << 16);        // tensor_dim0 hi=0; tensor_dim1=1
    g1[3] = (int)(8192u << 16);     // tensor_dim1 hi=0; tile_dim0=8192
    g1[4] = 1;                      // tile_dim1=1; tile_dim2=0
    g1[5] = 8192;                   // tensor_dim0_stride lo
    g1[6] = 0;
    g1[7] = 0;
    i32x4 z4 = {0, 0, 0, 0};
#if defined(TDM_6ARG)
    i32x8 z8 = {0, 0, 0, 0, 0, 0, 0, 0};
    __builtin_amdgcn_tensor_load_to_lds(g0, g1, z4, z4, z8, 0);
#else
    __builtin_amdgcn_tensor_load_to_lds(g0, g1, z4, z4, 0);
#endif
  }
#endif

  // gather x = bf16(emb[tok]) : 4 threads per batch row, 32 elems each
  {
    int b = tid >> 2, h0i = (tid & 3) * 32;
    int tok = tokens[(off + node) * 64 + b];
    const float* er = emb + (long long)tok * 128 + h0i;
    __bf16* xr = xL + b * LDA + h0i;
#pragma unroll
    for (int i = 0; i < 32; ++i) xr[i] = (__bf16)er[i];
  }

  if (has_children) {
#if defined(HAVE_TDM)
    if (wave == 0) __builtin_amdgcn_s_wait_tensorcnt(0);
#else
    // fallback: manual copy of 4 children hidden states into padded LDS rows
    {
      int r = tid;                               // one row per thread
      const uint4* src = (const uint4*)(hs + (size_t)(off_child + node * 4) * 8192);
      uint4* d = (uint4*)(chL + r * LDA);
      const uint4* s = src + r * 16;             // 128 bf16 = 16 uint4
#pragma unroll
      for (int i = 0; i < 16; ++i) d[i] = s[i];
    }
#endif
    __syncthreads();

    // ---- u = tanh(ch * sent_w + sent_b); s[row] += u * ctx_w (fused) ----
    for (int t = wave; t < 128; t += 8) {        // 16 Mtiles x 8 Ntiles
      int m0 = (t >> 3) * 16, n0 = (t & 7) * 16;
      const __bf16* arow = chL + (m0 + l) * LDA + hf * 16;
      const __bf16* brow = swT + (n0 + l) * 128 + hf * 16;
      v8f acc = {};
      acc = gemm16(arow, brow, acc);
      int col = n0 + l;
      float sb = sent_b[col], cw = ctx_w[col];
      int rbase = m0 + hf * 8;
#pragma unroll
      for (int v = 0; v < 8; ++v)
        atomicAdd(&sL[rbase + v], fast_tanh(acc[v] + sb) * cw);
    }
    __syncthreads();

    // ---- softmax over the 4 children (per batch lane b) ----
    if (tid < 64) {
      float s0 = fast_tanh(sL[tid]),       s1 = fast_tanh(sL[64 + tid]);
      float s2 = fast_tanh(sL[128 + tid]), s3 = fast_tanh(sL[192 + tid]);
      float m = fmaxf(fmaxf(s0, s1), fmaxf(s2, s3));
      float e0 = __expf(s0 - m), e1 = __expf(s1 - m);
      float e2 = __expf(s2 - m), e3 = __expf(s3 - m);
      float inv = __builtin_amdgcn_rcpf(e0 + e1 + e2 + e3);
      aL[tid] = e0 * inv; aL[64 + tid] = e1 * inv;
      aL[128 + tid] = e2 * inv; aL[192 + tid] = e3 * inv;
    }
    __syncthreads();

    // ---- h0 = sum_a alpha[a,b] * ch[a,b,:] ----
    {
      int b = tid >> 2, hb = (tid & 3) * 32;
      float a0 = aL[b], a1 = aL[64 + b], a2 = aL[128 + b], a3 = aL[192 + b];
      const __bf16* c0 = chL + (b)       * LDA + hb;
      const __bf16* c1 = chL + (64 + b)  * LDA + hb;
      const __bf16* c2 = chL + (128 + b) * LDA + hb;
      const __bf16* c3 = chL + (192 + b) * LDA + hb;
      __bf16* hr = h0L + b * LDA + hb;
#pragma unroll
      for (int i = 0; i < 32; ++i)
        hr[i] = (__bf16)(a0 * (float)c0[i] + a1 * (float)c1[i] +
                         a2 * (float)c2[i] + a3 * (float)c3[i]);
    }
  } else {
    int b = tid >> 2, hb = (tid & 3) * 32;
    __bf16* hr = h0L + b * LDA + hb;
#pragma unroll
    for (int i = 0; i < 32; ++i) hr[i] = (__bf16)0.0f;
  }
  __syncthreads();

  // ---- GRU: each unit = (m-tile, j) owns gate-column triple {j, j+8, j+16} ----
  __bf16* hout = hs + (size_t)(off + node) * 8192;
  for (int u = wave; u < 32; u += 8) {           // 4 Mtiles x 8 col-tiles
    int m0 = (u >> 3) * 16, j = u & 7;
    const __bf16* xrow = xL  + (m0 + l) * LDA + hf * 16;
    const __bf16* hrow = h0L + (m0 + l) * LDA + hf * 16;
    int cr = j * 16 + l, cz = cr + 128, cn = cr + 256;

    v8f ir = {}, hr = {}, iz = {}, hz = {}, in_ = {}, hn_ = {};
    ir  = gemm16(xrow, wih + cr * 128 + hf * 16, ir);
    hr  = gemm16(hrow, whh + cr * 128 + hf * 16, hr);
    iz  = gemm16(xrow, wih + cz * 128 + hf * 16, iz);
    hz  = gemm16(hrow, whh + cz * 128 + hf * 16, hz);
    in_ = gemm16(xrow, wih + cn * 128 + hf * 16, in_);
    hn_ = gemm16(hrow, whh + cn * 128 + hf * 16, hn_);

    float br  = b_ih[cr] + b_hh[cr];
    float bz  = b_ih[cz] + b_hh[cz];
    float bin = b_ih[cn], bhn = b_hh[cn];
    int hcol  = j * 16 + l;                      // hidden index of this lane
    int rbase = m0 + hf * 8;                     // batch-row base
#pragma unroll
    for (int v = 0; v < 8; ++v) {
      int b = rbase + v;
      float r  = fast_sigmoid(ir[v] + hr[v] + br);
      float z  = fast_sigmoid(iz[v] + hz[v] + bz);
      float n  = fast_tanh(in_[v] + bin + r * (hn_[v] + bhn));
      float hp = (float)h0L[b * LDA + hcol];
      float hv = (1.0f - z) * n + z * hp;
      hout[b * 128 + hcol] = (__bf16)hv;
      atomicMaxFloat(out + b * 128 + hcol, hv);
    }
  }
}

// ---------------------------------------------------------------------------
extern "C" void kernel_launch(void* const* d_in, const int* in_sizes, int n_in,
                              void* d_out, int out_size, void* d_ws, size_t ws_size,
                              hipStream_t stream) {
  const int*   tokens = (const int*)d_in[0];
  const float* emb    = (const float*)d_in[1];
  const float* sent_w = (const float*)d_in[2];
  const float* sent_b = (const float*)d_in[3];
  const float* ctx_w  = (const float*)d_in[4];
  const float* w_ih   = (const float*)d_in[5];
  const float* w_hh   = (const float*)d_in[6];
  const float* b_ih   = (const float*)d_in[7];
  const float* b_hh   = (const float*)d_in[8];
  float* out = (float*)d_out;

  // workspace: bf16 weights + full-tree bf16 hidden states (~89.7 MB)
  char* ws = (char*)d_ws;
  unsigned short* w_ih_b  = (unsigned short*)(ws);            // 384*128 bf16
  unsigned short* w_hh_b  = (unsigned short*)(ws + 98304);    // 384*128 bf16
  unsigned short* sent_wT = (unsigned short*)(ws + 196608);   // 128*128 bf16 (transposed)
  unsigned short* hs      = (unsigned short*)(ws + 229376);   // 5461*64*128 bf16

  (void)in_sizes; (void)n_in; (void)out_size; (void)ws_size;

  hipFuncSetAttribute((const void*)tree_gru_level,
                      hipFuncAttributeMaxDynamicSharedMemorySize, SMEM_BYTES);

  prep_kernel<<<192, 256, 0, stream>>>(w_ih, w_hh, sent_w,
                                       w_ih_b, w_hh_b, sent_wT, out);

  int n = 4096, off = 1365;                  // level d = 6 (leaves) upward
  for (int d = 6; d >= 0; --d) {
    int off_child = 4 * off + 1;
    tree_gru_level<<<n, 256, SMEM_BYTES, stream>>>(
        tokens, emb, sent_b, ctx_w, b_ih, b_hh,
        w_ih_b, w_hh_b, sent_wT, hs, out,
        off, off_child, (d < 6) ? 1 : 0);
    n >>= 2;
    off = (off - 1) >> 2;
  }
}
